// VANLayer_68453188764122
// MI455X (gfx1250) — matrix operand order repaired and checked by reading, not compiled, not run on hardware
//
#include <hip/hip_runtime.h>
#include <hip/hip_bf16.h>

// ---------------------------------------------------------------------------
// VAN / GraphGPS layer for MI455X (gfx1250, wave32, WMMA).
// Dense GEMMs -> v_wmma_f32_16x16x32_bf16 ; sparse softmax-aggregation -> L2
// float atomics ; BatchNorm -> two-pass block-reduce + atomics.
// ---------------------------------------------------------------------------

typedef __bf16 bf16_t;
typedef __attribute__((ext_vector_type(16))) __bf16 v16bf;
typedef __attribute__((ext_vector_type(8)))  __bf16 v8bf;
typedef __attribute__((ext_vector_type(4)))  __bf16 v4bf;
typedef __attribute__((ext_vector_type(8)))  float  v8f;

#define DIMH  128
#define HEADS 8
#define HD    16

// ---------------------------------------------------------------------------
// Generic bf16 WMMA GEMM:  out[M x Nout] = A[M x KT] @ W[Nout x KT]^T + bias
// One 16x16 output tile per wave; 8 waves per block cover 128 output columns.
// Fragment layouts used here (wave32):
//   A (16x32 bf16): lane L holds row L%16; 16 contiguous ks from 16*(L/16).
//   B (32x16 bf16): lane L holds col L%16; 16 contiguous ks from 16*(L/16).
//   C/D (16x16 f32): lane L col L%16; VGPR v -> row v + 8*(L/16).
// GUARD=false is selected at launch when M % 16 == 0 (true for N=40000):
// removes all EXEC-guard branches from the store epilogue.
// ---------------------------------------------------------------------------
template <int KT, bool RELU, bool ADD_RES, bool WRITE_F32, bool WRITE_BF16,
          bool GUARD>
__global__ __launch_bounds__(256) void gemm_bf16_wmma(
    const bf16_t* __restrict__ A, const bf16_t* __restrict__ W,
    const float* __restrict__ bias, const float* __restrict__ res,
    float* __restrict__ outF, bf16_t* __restrict__ outB, int M, int Nout) {
  const int lane = threadIdx.x & 31;
  const int wave = threadIdx.x >> 5;              // 0..7
  const int lmod = lane & 15;
  const int lhi  = lane >> 4;                     // 0 or 1

  const int rowBase = blockIdx.x * 16;
  const int colTile = wave + (blockIdx.y << 3);
  const int colBase = colTile * 16;

  int arow = rowBase + lmod;
  if (GUARD && arow > M - 1) arow = M - 1;        // clamp (epilogue guards)
  const bf16_t* __restrict__ ap = A + (size_t)arow * KT;
  const bf16_t* __restrict__ wp = W + (size_t)(colBase + lmod) * KT;

  v8f c = {};
#pragma unroll
  for (int k = 0; k < KT; k += 32) {
    v8bf a0 = *(const v8bf*)(ap + k + lhi * 16);
    v8bf a1 = *(const v8bf*)(ap + k + lhi * 16 + 8);
    v16bf a, b;
#pragma unroll
    for (int i = 0; i < 8; ++i) { a[i] = a0[i]; a[8 + i] = a1[i]; }
    b = *(const v16bf*)(wp + k + lhi * 16);
    c = __builtin_amdgcn_wmma_f32_16x16x32_bf16(false, a, false, b,
                                                (short)0, c, false, false);
  }

  const float bc = bias[colBase + lmod];
  const size_t idx0 =
      (size_t)(rowBase + (lhi << 3)) * Nout + (colBase + lmod);
#pragma unroll
  for (int v = 0; v < 8; ++v) {
    if (GUARD && (rowBase + v + (lhi << 3)) >= M) continue;
    const size_t idx = idx0 + (size_t)v * Nout;
    float val = c[v] + bc;
    if (ADD_RES) val += res[idx];
    if (RELU) val = val > 0.0f ? val : 0.0f;
    if (WRITE_F32) outF[idx] = val;
    if (WRITE_BF16) outB[idx] = (bf16_t)val;
  }
}

template <int KT, bool RELU, bool ADD_RES, bool WF, bool WB>
static void launch_gemm(dim3 grid, hipStream_t s, const bf16_t* A,
                        const bf16_t* W, const float* bias, const float* res,
                        float* outF, bf16_t* outB, int M, int Nout) {
  if ((M & 15) == 0)
    gemm_bf16_wmma<KT, RELU, ADD_RES, WF, WB, false>
        <<<grid, 256, 0, s>>>(A, W, bias, res, outF, outB, M, Nout);
  else
    gemm_bf16_wmma<KT, RELU, ADD_RES, WF, WB, true>
        <<<grid, 256, 0, s>>>(A, W, bias, res, outF, outB, M, Nout);
}

// ---------------------------------------------------------------------------
// Elementwise helpers (vectorized: every tensor length is a multiple of 128)
// ---------------------------------------------------------------------------
__global__ __launch_bounds__(256) void f32_to_bf16_kernel(
    const float* __restrict__ in, bf16_t* __restrict__ out, int n4) {
  int i = blockIdx.x * 256 + threadIdx.x;
  if (i >= n4) return;
  float4 f = ((const float4*)in)[i];
  v4bf o;
  o[0] = (bf16_t)f.x; o[1] = (bf16_t)f.y; o[2] = (bf16_t)f.z; o[3] = (bf16_t)f.w;
  ((v4bf*)out)[i] = o;
}

__global__ __launch_bounds__(256) void init_accum_kernel(
    float* __restrict__ m, float* __restrict__ d, float* __restrict__ agg,
    float* __restrict__ stats, int nm, int nagg) {
  int i = blockIdx.x * 256 + threadIdx.x;
  if (i < nm) { m[i] = -3.402823466e38f; d[i] = 0.0f; }
  if (i < nagg) agg[i] = 0.0f;
  if (i < 256) { stats[i] = 0.0f; stats[512 + i] = 0.0f; }  // sums1, sums2
}

// ---------------------------------------------------------------------------
// Edge phase: scores -> segment max -> exp/denom -> weighted aggregation.
// One thread per (edge, head). Node tables live in the 192MB L2; segment
// reductions use hardware float atomics (global_atomic_max_num_f32/add_f32).
// ---------------------------------------------------------------------------
__global__ __launch_bounds__(256) void edge_scores_kernel(
    const float* __restrict__ Kb, const float* __restrict__ x,
    const int* __restrict__ ei, float* __restrict__ S, float* __restrict__ m,
    int E) {
  int t = blockIdx.x * 256 + threadIdx.x;
  if (t >= E * HEADS) return;
  const int e = t >> 3, h = t & 7;
  const int src = ei[e], dst = ei[E + e];
  const float4* kp = (const float4*)(Kb + (size_t)src * DIMH + h * HD);
  const float4* qp = (const float4*)(x + (size_t)dst * DIMH + h * HD);
  float s = 0.0f;
#pragma unroll
  for (int i = 0; i < 4; ++i) {
    float4 a = kp[i], b = qp[i];
    s += a.x * b.x + a.y * b.y + a.z * b.z + a.w * b.w;
  }
  s *= 0.25f;  // 1/sqrt(HD)
  S[t] = s;
  atomicMax(&m[dst * HEADS + h], s);
}

__global__ __launch_bounds__(256) void edge_exp_kernel(
    const int* __restrict__ ei, float* __restrict__ S,
    const float* __restrict__ m, float* __restrict__ denom, int E) {
  int t = blockIdx.x * 256 + threadIdx.x;
  if (t >= E * HEADS) return;
  const int e = t >> 3, h = t & 7;
  const int dst = ei[E + e];
  float v = __expf(S[t] - m[dst * HEADS + h]);
  S[t] = v;
  atomicAdd(&denom[dst * HEADS + h], v);
}

__global__ __launch_bounds__(256) void edge_aggregate_kernel(
    const int* __restrict__ ei, const float* __restrict__ S,
    const float* __restrict__ denom, const float* __restrict__ V,
    float* __restrict__ agg, int E) {
  int t = blockIdx.x * 256 + threadIdx.x;
  if (t >= E * HEADS) return;
  const int e = t >> 3, h = t & 7;
  const int src = ei[e], dst = ei[E + e];
  const float attn = S[t] / (denom[dst * HEADS + h] + 1e-16f);
  const float* __restrict__ vp = V + (size_t)src * DIMH + h * HD;
  float* __restrict__ op = agg + (size_t)dst * DIMH + h * HD;
#pragma unroll
  for (int i = 0; i < HD; ++i) atomicAdd(op + i, vp[i] * attn);
}

// ---------------------------------------------------------------------------
// BatchNorm (training mode, batch stats over N rows, 128 features)
// ---------------------------------------------------------------------------
__global__ __launch_bounds__(128) void bn_stats_kernel(
    const float* __restrict__ h, float* __restrict__ sums, int Nrows) {
  const int col = threadIdx.x;  // 0..127
  const int r0 = blockIdx.x * 64;
  int r1 = r0 + 64; if (r1 > Nrows) r1 = Nrows;
  float s = 0.0f, s2 = 0.0f;
  for (int r = r0; r < r1; ++r) {
    float v = h[(size_t)r * DIMH + col];
    s += v; s2 += v * v;
  }
  atomicAdd(&sums[col], s);
  atomicAdd(&sums[DIMH + col], s2);
}

__global__ __launch_bounds__(128) void bn_finalize_kernel(
    const float* __restrict__ sums, const float* __restrict__ g,
    const float* __restrict__ b, float* __restrict__ sc, float invN) {
  const int c = threadIdx.x;
  const float mu = sums[c] * invN;
  const float var = sums[DIMH + c] * invN - mu * mu;
  const float s = rsqrtf(var + 1e-5f) * g[c];
  sc[c] = s;
  sc[DIMH + c] = b[c] - mu * s;
}

template <bool WRITE_BF16>
__global__ __launch_bounds__(256) void bn_apply_kernel(
    const float* __restrict__ in, const float* __restrict__ sc,
    float* __restrict__ outF, bf16_t* __restrict__ outB, int n4) {
  int i = blockIdx.x * 256 + threadIdx.x;
  if (i >= n4) return;
  const int c = (i << 2) & (DIMH - 1);
  const float4 f = ((const float4*)in)[i];
  float4 o;
  o.x = f.x * sc[c + 0] + sc[DIMH + c + 0];
  o.y = f.y * sc[c + 1] + sc[DIMH + c + 1];
  o.z = f.z * sc[c + 2] + sc[DIMH + c + 2];
  o.w = f.w * sc[c + 3] + sc[DIMH + c + 3];
  ((float4*)outF)[i] = o;
  if (WRITE_BF16) {
    v4bf b;
    b[0] = (bf16_t)o.x; b[1] = (bf16_t)o.y; b[2] = (bf16_t)o.z; b[3] = (bf16_t)o.w;
    ((v4bf*)outB)[i] = b;
  }
}

// ---------------------------------------------------------------------------
// Host-side orchestration
// ---------------------------------------------------------------------------
static inline size_t alignUp(size_t x) { return (x + 255) & ~(size_t)255; }

extern "C" void kernel_launch(void* const* d_in, const int* in_sizes, int n_in,
                              void* d_out, int out_size, void* d_ws,
                              size_t ws_size, hipStream_t stream) {
  const float* x    = (const float*)d_in[0];
  const int*   ei   = (const int*)d_in[1];
  const float* Wk   = (const float*)d_in[2];
  const float* bk   = (const float*)d_in[3];
  const float* Wv   = (const float*)d_in[4];
  const float* bv   = (const float*)d_in[5];
  const float* Wo   = (const float*)d_in[6];
  const float* bo   = (const float*)d_in[7];
  const float* bn1g = (const float*)d_in[8];
  const float* bn1b = (const float*)d_in[9];
  const float* W1   = (const float*)d_in[10];
  const float* b1   = (const float*)d_in[11];
  const float* W2   = (const float*)d_in[12];
  const float* b2   = (const float*)d_in[13];
  const float* bn2g = (const float*)d_in[14];
  const float* bn2b = (const float*)d_in[15];

  const int N = in_sizes[0] / DIMH;      // 40000
  const int E = in_sizes[1] / 2;         // 640000

  // ---- workspace layout (bytes) ----
  char* ws = (char*)d_ws;
  size_t off = 0;
  auto take = [&](size_t bytes) { char* p = ws + off; off += alignUp(bytes); return p; };

  bf16_t* xb   = (bf16_t*)take((size_t)N * DIMH * 2);  // reused as agg_bf16, h_bf16
  bf16_t* wkb  = (bf16_t*)take(DIMH * DIMH * 2);
  bf16_t* wvb  = (bf16_t*)take(DIMH * DIMH * 2);
  bf16_t* wob  = (bf16_t*)take(DIMH * DIMH * 2);
  bf16_t* w1b  = (bf16_t*)take(2 * DIMH * DIMH * 2);
  bf16_t* w2b  = (bf16_t*)take(2 * DIMH * DIMH * 2);
  float* Kbuf  = (float*)take((size_t)N * DIMH * 4);
  float* Vbuf  = (float*)take((size_t)N * DIMH * 4);
  float* Sbuf  = (float*)take((size_t)E * HEADS * 4);
  float* mbuf  = (float*)take((size_t)N * HEADS * 4);
  float* dbuf  = (float*)take((size_t)N * HEADS * 4);
  float* agg   = (float*)take((size_t)N * DIMH * 4);
  float* r1    = (float*)take((size_t)N * DIMH * 4);   // reused as r2
  float* hbuf  = (float*)take((size_t)N * DIMH * 4);
  bf16_t* tbf  = (bf16_t*)take((size_t)N * 2 * DIMH * 2);
  float* stats = (float*)take(1024 * 4);  // [0:256) sums1 [256:512) sc1 [512:768) sums2 [768:1024) sc2
  (void)ws_size;

  bf16_t* aggb = xb;  // alias (xb dead after K/V gemms)
  bf16_t* hb   = xb;  // alias (aggb dead after Wo gemm)

  const int nND  = N * DIMH;          // 5.12M
  const int nNH  = N * HEADS;
  const int nEH  = E * HEADS;
  const int rowT = (N + 15) / 16;     // 2500
  const int WW   = DIMH * DIMH;       // 16384

  // 1. init accumulators (every call: determinism under graph replay)
  init_accum_kernel<<<(nND + 255) / 256, 256, 0, stream>>>(mbuf, dbuf, agg, stats, nNH, nND);

  // 2. bf16 conversions (float4 -> 4x bf16 per thread)
  f32_to_bf16_kernel<<<(nND / 4 + 255) / 256, 256, 0, stream>>>(x, xb, nND / 4);
  f32_to_bf16_kernel<<<(WW / 4 + 255) / 256, 256, 0, stream>>>(Wk, wkb, WW / 4);
  f32_to_bf16_kernel<<<(WW / 4 + 255) / 256, 256, 0, stream>>>(Wv, wvb, WW / 4);
  f32_to_bf16_kernel<<<(WW / 4 + 255) / 256, 256, 0, stream>>>(Wo, wob, WW / 4);
  f32_to_bf16_kernel<<<(WW / 2 + 255) / 256, 256, 0, stream>>>(W1, w1b, WW / 2);
  f32_to_bf16_kernel<<<(WW / 2 + 255) / 256, 256, 0, stream>>>(W2, w2b, WW / 2);

  // 3. K = x@Wk^T + bk ; V = x@Wv^T + bv     (WMMA)
  launch_gemm<DIMH, false, false, true, false>(dim3(rowT, 1), stream,
      xb, wkb, bk, nullptr, Kbuf, nullptr, N, DIMH);
  launch_gemm<DIMH, false, false, true, false>(dim3(rowT, 1), stream,
      xb, wvb, bv, nullptr, Vbuf, nullptr, N, DIMH);

  // 4-6. sparse softmax attention over edges
  edge_scores_kernel<<<(nEH + 255) / 256, 256, 0, stream>>>(Kbuf, x, ei, Sbuf, mbuf, E);
  edge_exp_kernel<<<(nEH + 255) / 256, 256, 0, stream>>>(ei, Sbuf, mbuf, dbuf, E);
  edge_aggregate_kernel<<<(nEH + 255) / 256, 256, 0, stream>>>(ei, Sbuf, dbuf, Vbuf, agg, E);

  // 7-8. h_attn = agg@Wo^T + bo ;  r1 = x + h_attn   (WMMA, fused residual)
  f32_to_bf16_kernel<<<(nND / 4 + 255) / 256, 256, 0, stream>>>(agg, aggb, nND / 4);
  launch_gemm<DIMH, false, true, true, false>(dim3(rowT, 1), stream,
      aggb, wob, bo, x, r1, nullptr, N, DIMH);

  // 9-11. BatchNorm1 -> h (f32 + bf16)
  bn_stats_kernel<<<(N + 63) / 64, 128, 0, stream>>>(r1, stats, N);
  bn_finalize_kernel<<<1, 128, 0, stream>>>(stats, bn1g, bn1b, stats + 256, 1.0f / N);
  bn_apply_kernel<true><<<(nND / 4 + 255) / 256, 256, 0, stream>>>(r1, stats + 256, hbuf, hb, nND / 4);

  // 12. t = relu(h@W1^T + b1)   (N x 256, bf16 out)    (WMMA, fused ReLU)
  launch_gemm<DIMH, true, false, false, true>(dim3(rowT, 2), stream,
      hb, w1b, b1, nullptr, nullptr, tbf, N, 2 * DIMH);

  // 13. r2 = h + t@W2^T + b2    (WMMA, fused residual)
  launch_gemm<2 * DIMH, false, true, true, false>(dim3(rowT, 1), stream,
      tbf, w2b, b2, hbuf, r1, nullptr, N, DIMH);

  // 14-16. BatchNorm2 -> d_out
  bn_stats_kernel<<<(N + 63) / 64, 128, 0, stream>>>(r1, stats + 512, N);
  bn_finalize_kernel<<<1, 128, 0, stream>>>(stats + 512, bn2g, bn2b, stats + 768, 1.0f / N);
  bn_apply_kernel<false><<<(nND / 4 + 255) / 256, 256, 0, stream>>>(r1, stats + 768, (float*)d_out, nullptr, nND / 4);
}
